// RealNVP_Bridge_3L_8211977470758
// MI455X (gfx1250) — matrix operand order repaired and checked by reading, not compiled, not run on hardware
//
#include <hip/hip_runtime.h>
#include <hip/hip_bf16.h>

#define BATCH 16384
#define DLAT  256
#define HID   512
#define EMB   128
#define NCLS  16
#define NLAYER 8

typedef __attribute__((ext_vector_type(16))) _Float16 v16h;
typedef __attribute__((ext_vector_type(8)))  _Float16 v8h;
typedef __attribute__((ext_vector_type(2)))  _Float16 v2h;
typedef __attribute__((ext_vector_type(8)))  float    v8f;
typedef __attribute__((ext_vector_type(4)))  float    v4f;

// ---- WMMA fragment helpers (wave32, v_wmma_f32_16x16x32_f16) -------------
// A (16x32 f16): lane l holds row m=l%16. halves 0..15 -> K = hi*16 + h:
//   lanes 0-15 : K0..15
//   lanes16-31 : K16..31
__device__ __forceinline__ int frag_k_a(int h, int hi) {
    return h + (hi ? 16 : 0);
}

__device__ __forceinline__ v16h load_a_frag(const _Float16* As, int m, int hi) {
    v16h a;
#pragma unroll
    for (int h = 0; h < 16; ++h) a[h] = As[m * 32 + frag_k_a(h, hi)];
    return a;  // two contiguous 16B runs -> 2x ds_load_b128
}

// B tile stored COLUMN-MAJOR in LDS: Bs[n*32 + k]. Fragment mapping:
// lane l holds column n=l%16; halves 0..15 -> K = hi*16 + h (contiguous!)
__device__ __forceinline__ v16h load_b_frag_sw(const _Float16* Bs, int n, int hi) {
    return *(const v16h*)(Bs + n * 32 + hi * 16);  // 32B contiguous -> 2x ds_load_b128
}

// ---- LDS staging ---------------------------------------------------------
// A tile: 128 rows x 32 halves from f16 src [*, stride]
__device__ __forceinline__ void stage_a_f16(_Float16* As, const _Float16* A,
                                            int blockRow, int stride, int k0, int tid) {
    int row  = tid >> 1;
    int half = (tid & 1) * 16;
    const v8h* src = (const v8h*)(A + (size_t)(blockRow + row) * stride + k0 + half);
    v8h* dst = (v8h*)(As + row * 32 + half);
    dst[0] = src[0];
    dst[1] = src[1];
}

// A tile from f32 src (z), converted to f16
__device__ __forceinline__ void stage_a_f32(_Float16* As, const float* A,
                                            int blockRow, int stride, int k0, int tid) {
    int row  = tid >> 1;
    int half = (tid & 1) * 16;
    const float* src = A + (size_t)(blockRow + row) * stride + k0 + half;
#pragma unroll
    for (int i = 0; i < 16; i += 4) {
        v4f v = *(const v4f*)(src + i);
        As[row * 32 + half + i + 0] = (_Float16)v[0];
        As[row * 32 + half + i + 1] = (_Float16)v[1];
        As[row * 32 + half + i + 2] = (_Float16)v[2];
        As[row * 32 + half + i + 3] = (_Float16)v[3];
    }
}

// B tile: 32 k-rows x 64 cols from f32 weights [K, ldw] starting (k0, c0).
// Transpose into column-major LDS layout Bs[n*32 + k], packing (k,k+1) pairs
// so stores are 4B ds_store_b32 and fragment reads are fully contiguous.
__device__ __forceinline__ void stage_b_f32_sw(_Float16* Bs, const float* W,
                                               int ldw, int k0, int c0, int tid) {
    int kp = (tid >> 4) * 2;   // k pair base: 0,2,...,30
    int n0 = (tid & 15) * 4;   // col base:    0,4,...,60
    const float* r0 = W + (size_t)(k0 + kp) * ldw + c0 + n0;
    const float* r1 = r0 + ldw;
    v4f a = *(const v4f*)r0;
    v4f b = *(const v4f*)r1;
#pragma unroll
    for (int i = 0; i < 4; ++i) {
        v2h pk;
        pk[0] = (_Float16)a[i];
        pk[1] = (_Float16)b[i];
        *(v2h*)(Bs + (n0 + i) * 32 + kp) = pk;
    }
}

// ---- Fused GEMM + FiLM + ReLU (+residual), double-buffered LDS -----------
// Out[b, c] = relu( (A@W + bias) * (1 + yemb@F[:,c] + Fb[c])
//                   + yemb@F[:,H+c] + Fb[H+c] ) [+ Ares]
template <bool A_IS_F32, bool RESIDUAL>
__global__ __launch_bounds__(256) void gemm_film_kernel(
    const void* __restrict__ Aptr, int astride, int kBegin, int kEnd,
    const float* __restrict__ W, const float* __restrict__ bias,
    const float* __restrict__ F, const float* __restrict__ Fb,
    const _Float16* __restrict__ yemb,
    const _Float16* __restrict__ Ares,
    _Float16* __restrict__ Out)
{
    __shared__ __align__(64) _Float16 As[2][128 * 32];
    __shared__ __align__(64) _Float16 Bs[2][64 * 32];   // column-major [n][k]
    __shared__ __align__(64) _Float16 Bs2[2][64 * 32];  // column-major [n][k]

    const int tid  = threadIdx.x;
    const int wave = tid >> 5, lane = tid & 31;
    const int wm = wave & 3, wn = wave >> 2;
    const int hi = lane >> 4, lr = lane & 15;
    const int blockRow = blockIdx.x * 128;
    const int c0 = blockIdx.y * 64;

    v8f acc[2][2]  = {};
    v8f gacc[2][2] = {};
    v8f bacc[2][2] = {};

    // ---- main GEMM over (possibly mask-restricted) K range, double-buffered
    const int nk = (kEnd - kBegin) >> 5;
    int buf = 0;
    if (A_IS_F32) stage_a_f32(As[0], (const float*)Aptr, blockRow, astride, kBegin, tid);
    else          stage_a_f16(As[0], (const _Float16*)Aptr, blockRow, astride, kBegin, tid);
    stage_b_f32_sw(Bs[0], W, HID, kBegin, c0, tid);

    for (int i = 0; i < nk; ++i) {
        __syncthreads();
        if (i + 1 < nk) {
            const int kn = kBegin + (i + 1) * 32;
            if (A_IS_F32) stage_a_f32(As[buf ^ 1], (const float*)Aptr, blockRow, astride, kn, tid);
            else          stage_a_f16(As[buf ^ 1], (const _Float16*)Aptr, blockRow, astride, kn, tid);
            stage_b_f32_sw(Bs[buf ^ 1], W, HID, kn, c0, tid);
            if (i + 2 < nk)
                __builtin_prefetch(W + (size_t)(kn + 32 + (tid >> 4) * 2) * HID + c0, 0, 1);
        }
        v16h af[2], bf[2];
        af[0] = load_a_frag(As[buf], wm * 32 + lr, hi);
        af[1] = load_a_frag(As[buf], wm * 32 + 16 + lr, hi);
        bf[0] = load_b_frag_sw(Bs[buf], wn * 32 + lr, hi);
        bf[1] = load_b_frag_sw(Bs[buf], wn * 32 + 16 + lr, hi);
#pragma unroll
        for (int mi = 0; mi < 2; ++mi)
#pragma unroll
            for (int ni = 0; ni < 2; ++ni)
                acc[mi][ni] = __builtin_amdgcn_wmma_f32_16x16x32_f16(
                    false, af[mi], false, bf[ni], (short)0, acc[mi][ni], false, false);
        buf ^= 1;
    }

    // ---- FiLM GEMMs: gamma/beta = yemb @ F[:, c0..], F[:, HID+c0..] (K=128)
    // First stage targets As[buf]/Bs[buf]: the buffer the last WMMA did NOT read.
    stage_a_f16(As[buf], yemb, blockRow, EMB, 0, tid);
    stage_b_f32_sw(Bs[buf],  F, 2 * HID, 0, c0, tid);
    stage_b_f32_sw(Bs2[buf], F, 2 * HID, 0, HID + c0, tid);

    for (int i = 0; i < EMB / 32; ++i) {
        __syncthreads();
        if (i + 1 < EMB / 32) {
            const int kn = (i + 1) * 32;
            stage_a_f16(As[buf ^ 1], yemb, blockRow, EMB, kn, tid);
            stage_b_f32_sw(Bs[buf ^ 1],  F, 2 * HID, kn, c0, tid);
            stage_b_f32_sw(Bs2[buf ^ 1], F, 2 * HID, kn, HID + c0, tid);
        }
        v16h af[2], bg[2], bb[2];
        af[0] = load_a_frag(As[buf], wm * 32 + lr, hi);
        af[1] = load_a_frag(As[buf], wm * 32 + 16 + lr, hi);
        bg[0] = load_b_frag_sw(Bs[buf],  wn * 32 + lr, hi);
        bg[1] = load_b_frag_sw(Bs[buf],  wn * 32 + 16 + lr, hi);
        bb[0] = load_b_frag_sw(Bs2[buf], wn * 32 + lr, hi);
        bb[1] = load_b_frag_sw(Bs2[buf], wn * 32 + 16 + lr, hi);
#pragma unroll
        for (int mi = 0; mi < 2; ++mi)
#pragma unroll
            for (int ni = 0; ni < 2; ++ni) {
                gacc[mi][ni] = __builtin_amdgcn_wmma_f32_16x16x32_f16(
                    false, af[mi], false, bg[ni], (short)0, gacc[mi][ni], false, false);
                bacc[mi][ni] = __builtin_amdgcn_wmma_f32_16x16x32_f16(
                    false, af[mi], false, bb[ni], (short)0, bacc[mi][ni], false, false);
            }
        buf ^= 1;
    }

    // ---- epilogue: C/D layout -> row = base + hi*8 + r, col = base + lane%16
#pragma unroll
    for (int mi = 0; mi < 2; ++mi)
#pragma unroll
        for (int ni = 0; ni < 2; ++ni)
#pragma unroll
            for (int r = 0; r < 8; ++r) {
                int row = blockRow + wm * 32 + mi * 16 + hi * 8 + r;
                int col = c0 + wn * 32 + ni * 16 + lr;
                float lin = acc[mi][ni][r] + bias[col];
                float g   = gacc[mi][ni][r] + Fb[col];
                float be  = bacc[mi][ni][r] + Fb[HID + col];
                float v = lin * (1.0f + g) + be;
                v = v > 0.0f ? v : 0.0f;
                if (RESIDUAL) v += (float)Ares[(size_t)row * HID + col];
                Out[(size_t)row * HID + col] = (_Float16)v;
            }
}

// ---- Output projection: Out(f32)[B,256] = A(f16)[B,512] @ Wo + bo --------
__global__ __launch_bounds__(256) void gemm_out_kernel(
    const _Float16* __restrict__ Ah, const float* __restrict__ Wo,
    const float* __restrict__ bo, float* __restrict__ Out)
{
    __shared__ __align__(64) _Float16 As[2][128 * 32];
    __shared__ __align__(64) _Float16 Bs[2][64 * 32];  // column-major [n][k]

    const int tid  = threadIdx.x;
    const int wave = tid >> 5, lane = tid & 31;
    const int wm = wave & 3, wn = wave >> 2;
    const int hi = lane >> 4, lr = lane & 15;
    const int blockRow = blockIdx.x * 128;
    const int c0 = blockIdx.y * 64;

    v8f acc[2][2] = {};
    const int nk = HID / 32;
    int buf = 0;
    stage_a_f16(As[0], Ah, blockRow, HID, 0, tid);
    stage_b_f32_sw(Bs[0], Wo, DLAT, 0, c0, tid);

    for (int i = 0; i < nk; ++i) {
        __syncthreads();
        if (i + 1 < nk) {
            const int kn = (i + 1) * 32;
            stage_a_f16(As[buf ^ 1], Ah, blockRow, HID, kn, tid);
            stage_b_f32_sw(Bs[buf ^ 1], Wo, DLAT, kn, c0, tid);
            if (i + 2 < nk)
                __builtin_prefetch(Wo + (size_t)(kn + 32 + (tid >> 4) * 2) * DLAT + c0, 0, 1);
        }
        v16h af[2], bf[2];
        af[0] = load_a_frag(As[buf], wm * 32 + lr, hi);
        af[1] = load_a_frag(As[buf], wm * 32 + 16 + lr, hi);
        bf[0] = load_b_frag_sw(Bs[buf], wn * 32 + lr, hi);
        bf[1] = load_b_frag_sw(Bs[buf], wn * 32 + 16 + lr, hi);
#pragma unroll
        for (int mi = 0; mi < 2; ++mi)
#pragma unroll
            for (int ni = 0; ni < 2; ++ni)
                acc[mi][ni] = __builtin_amdgcn_wmma_f32_16x16x32_f16(
                    false, af[mi], false, bf[ni], (short)0, acc[mi][ni], false, false);
        buf ^= 1;
    }
#pragma unroll
    for (int mi = 0; mi < 2; ++mi)
#pragma unroll
        for (int ni = 0; ni < 2; ++ni)
#pragma unroll
            for (int r = 0; r < 8; ++r) {
                int row = blockRow + wm * 32 + mi * 16 + hi * 8 + r;
                int col = c0 + wn * 32 + ni * 16 + lr;
                Out[(size_t)row * DLAT + col] = acc[mi][ni][r] + bo[col];
            }
}

// ---- Label embedding MLP: 16 -> 64(swish) -> 128(swish) -> 128 -----------
__global__ __launch_bounds__(256) void yemb_kernel(
    const float* __restrict__ y,
    const float* __restrict__ W1, const float* __restrict__ b1,
    const float* __restrict__ W2, const float* __restrict__ b2,
    const float* __restrict__ W3, const float* __restrict__ b3,
    _Float16* __restrict__ yemb)
{
    __shared__ float yrow[8][16];
    __shared__ float t1[8][64];
    __shared__ float t2[8][128];
    const int wave = threadIdx.x >> 5, lane = threadIdx.x & 31;
    const int row = blockIdx.x * 8 + wave;
    if (lane < 16) yrow[wave][lane] = y[(size_t)row * NCLS + lane];
    __syncthreads();
#pragma unroll
    for (int j = 0; j < 2; ++j) {
        int o = lane * 2 + j;
        float a = b1[o];
        for (int k = 0; k < 16; ++k) a += yrow[wave][k] * W1[k * 64 + o];
        t1[wave][o] = a / (1.0f + __expf(-a));
    }
    __syncthreads();
#pragma unroll
    for (int j = 0; j < 4; ++j) {
        int o = lane * 4 + j;
        float a = b2[o];
        for (int k = 0; k < 64; ++k) a += t1[wave][k] * W2[k * 128 + o];
        t2[wave][o] = a / (1.0f + __expf(-a));
    }
    __syncthreads();
#pragma unroll
    for (int j = 0; j < 4; ++j) {
        int o = lane * 4 + j;
        float a = b3[o];
        for (int k = 0; k < 128; ++k) a += t2[wave][k] * W3[k * 128 + o];
        yemb[(size_t)row * EMB + o] = (_Float16)a;
    }
}

// ---- Coupling update + log-det row reduction -----------------------------
__global__ __launch_bounds__(256) void coupling_update_kernel(
    float* __restrict__ z, const float* __restrict__ tout,
    const float* __restrict__ sout, float* __restrict__ ld, int maskFirstHalf)
{
    const int r = blockIdx.x, c = threadIdx.x;
    const size_t idx = (size_t)r * DLAT + c;
    float zv = z[idx];
    float s  = tanhf(sout[idx]);
    const bool keep = maskFirstHalf ? (c < DLAT / 2) : (c >= DLAT / 2);
    float contrib = 0.0f;
    if (!keep) {
        zv = zv * __expf(s) + tout[idx];
        contrib = s;
    }
    z[idx] = zv;
    __shared__ float red[256];
    red[c] = contrib;
    __syncthreads();
    for (int off = 128; off > 0; off >>= 1) {
        if (c < off) red[c] += red[c + off];
        __syncthreads();
    }
    if (c == 0) ld[r] += red[0];
}

__global__ void init_kernel(const float* __restrict__ x, float* __restrict__ z,
                            float* __restrict__ ld, int n)
{
    int i = blockIdx.x * blockDim.x + threadIdx.x;
    if (i < n) z[i] = x[i];
    if (i < BATCH) ld[i] = 0.0f;
}

// --------------------------------------------------------------------------
extern "C" void kernel_launch(void* const* d_in, const int* in_sizes, int n_in,
                              void* d_out, int out_size, void* d_ws, size_t ws_size,
                              hipStream_t stream)
{
    (void)in_sizes; (void)n_in; (void)out_size; (void)ws_size;
    const float* x = (const float*)d_in[0];
    const float* y = (const float*)d_in[1];

    float* z  = (float*)d_out;
    float* ld = z + (size_t)BATCH * DLAT;

    char* ws = (char*)d_ws;
    _Float16* yemb = (_Float16*)ws;
    _Float16* hA = (_Float16*)(ws + (size_t)BATCH * EMB * 2);
    _Float16* hB = (_Float16*)(ws + (size_t)BATCH * EMB * 2 + (size_t)BATCH * HID * 2);
    float* tout = (float*)(ws + (size_t)BATCH * EMB * 2 + 2 * (size_t)BATCH * HID * 2);
    float* sout = tout + (size_t)BATCH * DLAT;

    init_kernel<<<(BATCH * DLAT + 255) / 256, 256, 0, stream>>>(x, z, ld, BATCH * DLAT);

    for (int l = 0; l < NLAYER; ++l) {
        const float* eW1 = (const float*)d_in[2] + (size_t)l * NCLS * 64;
        const float* eb1 = (const float*)d_in[3] + (size_t)l * 64;
        const float* eW2 = (const float*)d_in[4] + (size_t)l * 64 * EMB;
        const float* eb2 = (const float*)d_in[5] + (size_t)l * EMB;
        const float* eW3 = (const float*)d_in[6] + (size_t)l * EMB * EMB;
        const float* eb3 = (const float*)d_in[7] + (size_t)l * EMB;
        yemb_kernel<<<BATCH / 8, 256, 0, stream>>>(y, eW1, eb1, eW2, eb2, eW3, eb3, yemb);

        const int kBegin = (l % 2 == 0) ? 0 : DLAT / 2;  // masked-K restriction
        const int kEnd   = kBegin + DLAT / 2;

        for (int br = 0; br < 2; ++br) {
            const int base = 8 + br * 14;  // t_ params then s_ params
            const float* W1  = (const float*)d_in[base + 0]  + (size_t)l * DLAT * HID;
            const float* b1  = (const float*)d_in[base + 1]  + (size_t)l * HID;
            const float* F1  = (const float*)d_in[base + 2]  + (size_t)l * EMB * 2 * HID;
            const float* Fb1 = (const float*)d_in[base + 3]  + (size_t)l * 2 * HID;
            const float* W2  = (const float*)d_in[base + 4]  + (size_t)l * HID * HID;
            const float* b2  = (const float*)d_in[base + 5]  + (size_t)l * HID;
            const float* F2  = (const float*)d_in[base + 6]  + (size_t)l * EMB * 2 * HID;
            const float* Fb2 = (const float*)d_in[base + 7]  + (size_t)l * 2 * HID;
            const float* W3  = (const float*)d_in[base + 8]  + (size_t)l * HID * HID;
            const float* b3  = (const float*)d_in[base + 9]  + (size_t)l * HID;
            const float* F3  = (const float*)d_in[base + 10] + (size_t)l * EMB * 2 * HID;
            const float* Fb3 = (const float*)d_in[base + 11] + (size_t)l * 2 * HID;
            const float* Wo  = (const float*)d_in[base + 12] + (size_t)l * HID * DLAT;
            const float* bo  = (const float*)d_in[base + 13] + (size_t)l * DLAT;

            dim3 g1(BATCH / 128, HID / 64);
            gemm_film_kernel<true, false><<<g1, 256, 0, stream>>>(
                z, DLAT, kBegin, kEnd, W1, b1, F1, Fb1, yemb, nullptr, hA);
            gemm_film_kernel<false, true><<<g1, 256, 0, stream>>>(
                hA, HID, 0, HID, W2, b2, F2, Fb2, yemb, hA, hB);
            gemm_film_kernel<false, true><<<g1, 256, 0, stream>>>(
                hB, HID, 0, HID, W3, b3, F3, Fb3, yemb, hB, hA);
            dim3 g2(BATCH / 128, DLAT / 64);
            gemm_out_kernel<<<g2, 256, 0, stream>>>(hA, Wo, bo, br == 0 ? tout : sout);
        }
        coupling_update_kernel<<<BATCH, 256, 0, stream>>>(z, tout, sout, ld,
                                                          (l % 2 == 0) ? 1 : 0);
    }
}
